// SpectralConv_70428873720142
// MI455X (gfx1250) — matrix operand order repaired and checked by reading, not compiled, not run on hardware
//
#include <hip/hip_runtime.h>

typedef __attribute__((ext_vector_type(16))) _Float16 v16h;
typedef __attribute__((ext_vector_type(8)))  _Float16 v8h;
typedef __attribute__((ext_vector_type(8)))  float    v8f;
typedef _Float16 h16;

#define WMMA16(a, b, c) \
  __builtin_amdgcn_wmma_f32_16x16x32_f16(false, (a), false, (b), (short)0, (c), false, false)

namespace {
constexpr int   kM1 = 20, kM2 = 20;
constexpr int   kNKX = 2 * kM1;          // 40 kx modes
constexpr int   kNPOS = kNKX * kM2;      // 800 mode positions
constexpr int   kROWS = 16 * 64 * 256;   // 262144 (B*Ci*H)
constexpr float kTW = 0.024543692606f;   // 2*pi/256

// ---- workspace layout (units: halves; every section 32B aligned) ----
constexpr size_t OFF_FWP = 0;                       // B-packed [8kt][3nt][32][16]  (256x48 fwd-W DFT)
constexpr size_t SZ_FWP  = 8ul * 3 * 32 * 16;
constexpr size_t OFF_CP  = OFF_FWP + SZ_FWP;        // A-packed [3mt][8kt][32][16]  cos (48x256 fwd-H)
constexpr size_t SZ_AH   = 3ul * 8 * 32 * 16;
constexpr size_t OFF_SP  = OFF_CP + SZ_AH;          // sin
constexpr size_t OFF_SNP = OFF_SP + SZ_AH;          // -sin
constexpr size_t OFF_GP  = OFF_SNP + SZ_AH;         // A-packed [16mt][3kt][32][16] (256x96 inv-H, /256)
constexpr size_t SZ_GP   = 16ul * 3 * 32 * 16;
constexpr size_t OFF_EP  = OFF_GP + SZ_GP;          // B-packed [2kt][16nt][32][16] (64x256 inv-W, /256)
constexpr size_t SZ_EP   = 2ul * 16 * 32 * 16;
constexpr size_t OFF_BWP = OFF_EP + SZ_EP;          // B-packed per p: [800][4kt][8nt][32][16]
constexpr size_t SZ_BWP  = 800ul * 4 * 8 * 32 * 16;
constexpr size_t OFF_XW  = OFF_BWP + SZ_BWP;        // [262144 rows][48]  (re/im interleaved, cols>=40 zero)
constexpr size_t SZ_XW   = 262144ul * 48;
constexpr size_t OFF_YP  = OFF_XW + SZ_XW;          // A-packed per p: [800][4kt][32][16] (16b x 128ci')
constexpr size_t SZ_YP   = 800ul * 4 * 32 * 16;
constexpr size_t OFF_OB  = OFF_YP + SZ_YP;          // [800][16 b][128 co'] plain
constexpr size_t SZ_OB   = 800ul * 16 * 128;
constexpr size_t OFF_ZB  = OFF_OB + SZ_OB;          // [1024 img][256 h][48]
constexpr size_t SZ_ZB   = 1024ul * 256 * 48;
}

// ---------- fragment helpers (documented CDNA5 wave32 WMMA layouts) ----------
// A 16x32 f16: lane l -> row m=l&15, kg=l>>4; half j -> k = (j&7) + ((j>>3)<<4) + 8*kg
// B 32x16 f16: lane l -> col n=l&15, kg=l>>4; half j -> k = (j&7) + ((j>>3)<<4) + 8*kg
// C/D 16x16 f32: elem r of lane l -> row = r + 8*(l>>4), col = l&15

__device__ __forceinline__ v16h load_a_rm(const h16* base, int ld, int lane) {
  int m = lane & 15, kg = lane >> 4;
  const h16* p = base + m * ld + kg * 8;
  v8h lo = *(const v8h*)p;
  v8h hi = *(const v8h*)(p + 16);
  v16h a;
#pragma unroll
  for (int j = 0; j < 8; ++j) { a[j] = lo[j]; a[j + 8] = hi[j]; }
  return a;
}

__device__ __forceinline__ size_t fragA_addr(int p, int k, int m) {
  int kt = k >> 5, kk = k & 31;
  int kg = (kk >> 3) & 1;
  int j  = (kk & 7) + ((kk >> 4) << 3);
  int lane = m + (kg << 4);
  return (((size_t)p * 4 + kt) * 32 + lane) * 16 + j;
}

// ======================= init: DFT factor matrices =======================
__global__ void k_init_fw(h16* ws) {            // 12288 threads
  int idx = blockIdx.x * 256 + threadIdx.x;
  int j = idx & 15, lane = (idx >> 4) & 31, ft = idx >> 9;   // ft = kt*3+nt
  int kk = (j & 7) + ((j >> 3) << 4) + ((lane >> 4) << 3);
  int w  = (ft / 3) * 32 + kk;
  int n  = (ft % 3) * 16 + (lane & 15);
  float v = 0.f;
  if (n < 2 * kM2) {
    int ky = n >> 1, s = n & 1;
    float ang = (float)((w * ky) & 255) * kTW;
    v = s ? -__sinf(ang) : __cosf(ang);
  }
  ws[OFF_FWP + idx] = (h16)v;
}

__global__ void k_init_h(h16* ws) {             // 3*12288 threads: C, S, -S
  int tid = blockIdx.x * 256 + threadIdx.x;
  int mat = tid / 12288, idx = tid % 12288;
  int j = idx & 15, lane = (idx >> 4) & 31, ft = idx >> 9;   // ft = mt*8+kt
  int m  = (ft >> 3) * 16 + (lane & 15);
  int kk = (j & 7) + ((j >> 3) << 4) + ((lane >> 4) << 3);
  int hh = (ft & 7) * 32 + kk;
  float v = 0.f;
  if (m < kNKX) {
    int kx = (m < kM1) ? m : (216 + m);
    float ang = (float)((hh * kx) & 255) * kTW;
    v = (mat == 0) ? __cosf(ang) : ((mat == 1) ? __sinf(ang) : -__sinf(ang));
  }
  ws[(mat == 0 ? OFF_CP : (mat == 1 ? OFF_SP : OFF_SNP)) + idx] = (h16)v;
}

__global__ void k_init_g(h16* ws) {             // 24576 threads (inv-H, 1/256 folded)
  int idx = blockIdx.x * 256 + threadIdx.x;
  int j = idx & 15, lane = (idx >> 4) & 31, ft = idx >> 9;   // ft = mt*3+kt
  int hh = (ft / 3) * 16 + (lane & 15);
  int kk = (j & 7) + ((j >> 3) << 4) + ((lane >> 4) << 3);
  int k  = (ft % 3) * 32 + kk;
  float v = 0.f;
  if (k < 2 * kNKX) {
    int kxi = k >> 1, s = k & 1;
    int kx = (kxi < kM1) ? kxi : (216 + kxi);
    float ang = (float)((hh * kx) & 255) * kTW;
    v = (s ? __sinf(ang) : __cosf(ang)) * (1.0f / 256.0f);
  }
  ws[OFF_GP + idx] = (h16)v;
}

__global__ void k_init_e(h16* ws) {             // 16384 threads (inv-W real, 1/256 + doubling)
  int idx = blockIdx.x * 256 + threadIdx.x;
  int j = idx & 15, lane = (idx >> 4) & 31, ft = idx >> 9;   // ft = kt*16+nt
  int kk = (j & 7) + ((j >> 3) << 4) + ((lane >> 4) << 3);
  int k  = (ft >> 4) * 32 + kk;
  int w  = (ft & 15) * 16 + (lane & 15);
  float v = 0.f;
  if (k < 2 * kM2) {
    int ky = k >> 1, s = k & 1;
    float amp = (ky == 0 ? 1.0f : 2.0f) * (1.0f / 256.0f);
    float ang = (float)((w * ky) & 255) * kTW;
    v = s ? -amp * __sinf(ang) : amp * __cosf(ang);
  }
  ws[OFF_EP + idx] = (h16)v;
}

// real-ified complex weights, B-packed per mode position
__global__ void k_init_bw(h16* ws, const float* __restrict__ w1r, const float* __restrict__ w1i,
                          const float* __restrict__ w2r, const float* __restrict__ w2i) {
  size_t tid = (size_t)blockIdx.x * 256 + threadIdx.x;      // 800*16384
  int p = (int)(tid >> 14);
  int rr = (int)(tid & 16383);
  int j = rr & 15, lane = (rr >> 4) & 31, ft = rr >> 9;     // ft = kt*8+nt
  int kk = (j & 7) + ((j >> 3) << 4) + ((lane >> 4) << 3);
  int k  = (ft >> 3) * 32 + kk;                             // 2*ci + s
  int n  = (ft & 7) * 16 + (lane & 15);                     // 2*co + t
  int ci = k >> 1, s = k & 1, co = n >> 1, t = n & 1;
  int kxi = p / kM2, ky = p % kM2;
  int km = (kxi < kM1) ? kxi : (kxi - kM1);
  const float* wr = (kxi < kM1) ? w1r : w2r;
  const float* wi = (kxi < kM1) ? w1i : w2i;
  size_t widx = (((size_t)ci * 64 + co) * kM1 + km) * kM2 + ky;
  float Wr = wr[widx], Wi = wi[widx];
  float v = (s == 0) ? (t == 0 ? Wr : Wi) : (t == 0 ? -Wi : Wr);
  ws[OFF_BWP + tid] = (h16)v;
}

// ======================= stage A: W-axis forward pruned DFT =======================
// [262144 x 256] x Fw[256 x 48] -> Xw f16, 64 rows per block (4 waves x 16 rows)
__global__ void __launch_bounds__(128) k_stageA(const float* __restrict__ x, h16* ws) {
  __shared__ h16 xs[64 * 264];
  int tid = threadIdx.x;
  int r0 = blockIdx.x * 64;
  const float4* xv = (const float4*)(x + (size_t)r0 * 256);
#pragma unroll
  for (int i = 0; i < 32; ++i) {
    int f4 = i * 128 + tid;
    float4 v = xv[f4];
    h16* d = &xs[(f4 >> 6) * 264 + (f4 & 63) * 4];
    d[0] = (h16)v.x; d[1] = (h16)v.y; d[2] = (h16)v.z; d[3] = (h16)v.w;
  }
  __syncthreads();
  int wave = tid >> 5, lane = tid & 31;
  v8f acc[3] = {};
  const h16* fw = ws + OFF_FWP;
  for (int kt = 0; kt < 8; ++kt) {
    v16h a = load_a_rm(&xs[(wave * 16) * 264 + kt * 32], 264, lane);
#pragma unroll
    for (int nt = 0; nt < 3; ++nt) {
      v16h b = *(const v16h*)(fw + ((size_t)(kt * 3 + nt) * 32 + lane) * 16);
      acc[nt] = WMMA16(a, b, acc[nt]);
    }
  }
  h16* xw = ws + OFF_XW;
  int kg = lane >> 4, n0 = lane & 15;
#pragma unroll
  for (int nt = 0; nt < 3; ++nt)
#pragma unroll
    for (int r = 0; r < 8; ++r) {
      int row = r0 + wave * 16 + r + 8 * kg;
      xw[(size_t)row * 48 + nt * 16 + n0] = (h16)acc[nt][r];
    }
}

// ======================= stage B: H-axis forward DFT per (b,ci) image =======================
__global__ void __launch_bounds__(128) k_stageB(h16* ws) {
  __shared__ h16 bx[16384];     // Xr packed [8kt][2nt][32][16] then Xi packed
  int tid = threadIdx.x;
  int img = blockIdx.x;
  int b = img >> 6, ci = img & 63;
  const h16* xw = ws + OFF_XW + (size_t)img * 256 * 48;
  for (int i = tid; i < 16384; i += 128) {
    int mat = i >> 13, r = i & 8191;
    int j = r & 15, lane = (r >> 4) & 31, ft = r >> 9;   // ft = kt*2+nt
    int kk = (j & 7) + ((j >> 3) << 4) + ((lane >> 4) << 3);
    int hh = (ft >> 1) * 32 + kk;
    int ky = (ft & 1) * 16 + (lane & 15);
    float v = 0.f;
    if (ky < kM2) v = (float)xw[hh * 48 + 2 * ky + mat];
    bx[i] = (h16)v;
  }
  __syncthreads();
  int wave = tid >> 5, lane = tid & 31;
  if (wave < 3) {
    v8f yre[2] = {}, yim[2] = {};
    const h16 *Cp = ws + OFF_CP, *Sp = ws + OFF_SP, *Np = ws + OFF_SNP;
    for (int kt = 0; kt < 8; ++kt) {
      size_t aoff = ((size_t)(wave * 8 + kt) * 32 + lane) * 16;
      v16h aC = *(const v16h*)(Cp + aoff);
      v16h aS = *(const v16h*)(Sp + aoff);
      v16h aN = *(const v16h*)(Np + aoff);
#pragma unroll
      for (int nt = 0; nt < 2; ++nt) {
        v16h br = *(const v16h*)(&bx[((kt * 2 + nt) * 32 + lane) * 16]);
        v16h bi = *(const v16h*)(&bx[8192 + ((kt * 2 + nt) * 32 + lane) * 16]);
        yre[nt] = WMMA16(aC, br, yre[nt]);
        yre[nt] = WMMA16(aS, bi, yre[nt]);
        yim[nt] = WMMA16(aC, bi, yim[nt]);
        yim[nt] = WMMA16(aN, br, yim[nt]);
      }
    }
    h16* yp = ws + OFF_YP;
    int kg = lane >> 4, n0 = lane & 15;
#pragma unroll
    for (int nt = 0; nt < 2; ++nt)
#pragma unroll
      for (int r = 0; r < 8; ++r) {
        int kxi = wave * 16 + r + 8 * kg;
        int ky = nt * 16 + n0;
        if (kxi < kNKX && ky < kM2) {
          int p = kxi * kM2 + ky;
          yp[fragA_addr(p, 2 * ci,     b)] = (h16)yre[nt][r];
          yp[fragA_addr(p, 2 * ci + 1, b)] = (h16)yim[nt][r];
        }
      }
  }
}

// ======================= stage C: complex channel mix per mode position =======================
__global__ void __launch_bounds__(128) k_stageC(h16* ws) {
  int tid = threadIdx.x, p = blockIdx.x;
  int wave = tid >> 5, lane = tid & 31;
  const h16* yp = ws + OFF_YP + (size_t)p * 2048;
  const h16* bw = ws + OFF_BWP + (size_t)p * 16384;
  v8f acc[2] = {};
  for (int kt = 0; kt < 4; ++kt) {
    v16h a = *(const v16h*)(yp + ((size_t)kt * 32 + lane) * 16);
#pragma unroll
    for (int q = 0; q < 2; ++q) {
      int nt = wave * 2 + q;
      v16h b = *(const v16h*)(bw + ((size_t)(kt * 8 + nt) * 32 + lane) * 16);
      acc[q] = WMMA16(a, b, acc[q]);
    }
  }
  h16* ob = ws + OFF_OB + (size_t)p * 2048;
  int kg = lane >> 4, n0 = lane & 15;
#pragma unroll
  for (int q = 0; q < 2; ++q)
#pragma unroll
    for (int r = 0; r < 8; ++r)
      ob[(r + 8 * kg) * 128 + (wave * 2 + q) * 16 + n0] = (h16)acc[q][r];
}

// ======================= stage D: inverse H DFT per (b,co) image =======================
__global__ void __launch_bounds__(128) k_stageD(h16* ws) {
  __shared__ h16 bp[4608];      // B' packed [3kt][3nt][32][16], K'=96, N'=48
  int tid = threadIdx.x;
  int img = blockIdx.x;
  int b = img >> 6, co = img & 63;
  const h16* ob = ws + OFF_OB;
  for (int i = tid; i < 4608; i += 128) {
    int j = i & 15, lane = (i >> 4) & 31, ft = i >> 9;   // ft = kt*3+nt
    int kk = (j & 7) + ((j >> 3) << 4) + ((lane >> 4) << 3);
    int k = (ft / 3) * 32 + kk;
    int n = (ft % 3) * 16 + (lane & 15);
    float v = 0.f;
    if (k < 2 * kNKX && n < 2 * kM2) {
      int kxi = k >> 1, s = k & 1, ky = n >> 1, t = n & 1;
      size_t obase = (size_t)(kxi * kM2 + ky) * 2048 + b * 128 + 2 * co;
      float Or = (float)ob[obase], Oi = (float)ob[obase + 1];
      v = (s == 0) ? (t == 0 ? Or : Oi) : (t == 0 ? -Oi : Or);
    }
    bp[i] = (h16)v;
  }
  __syncthreads();
  int wave = tid >> 5, lane = tid & 31;
  const h16* gp = ws + OFF_GP;
  h16* z = ws + OFF_ZB + (size_t)img * 256 * 48;
  for (int mi = 0; mi < 4; ++mi) {
    int mt = wave + mi * 4;
    v8f acc[3] = {};
    for (int kt = 0; kt < 3; ++kt) {
      v16h a = *(const v16h*)(gp + ((size_t)(mt * 3 + kt) * 32 + lane) * 16);
#pragma unroll
      for (int nt = 0; nt < 3; ++nt) {
        v16h bb = *(const v16h*)(&bp[((kt * 3 + nt) * 32 + lane) * 16]);
        acc[nt] = WMMA16(a, bb, acc[nt]);
      }
    }
    int kg = lane >> 4, n0 = lane & 15;
#pragma unroll
    for (int nt = 0; nt < 3; ++nt)
#pragma unroll
      for (int r = 0; r < 8; ++r) {
        int hh = mt * 16 + r + 8 * kg;
        z[hh * 48 + nt * 16 + n0] = (h16)acc[nt][r];
      }
  }
}

// ======================= stage E: inverse real W DFT, writes final output =======================
__global__ void __launch_bounds__(128) k_stageE(const h16* __restrict__ wsin, float* __restrict__ out) {
  __shared__ h16 zs[16 * 72];
  int tid = threadIdx.x;
  int row0 = blockIdx.x * 16;
  const h16* z = wsin + OFF_ZB + (size_t)row0 * 48;
  for (int i = tid; i < 16 * 64; i += 128) {
    int row = i >> 6, col = i & 63;
    zs[row * 72 + col] = (col < 48) ? z[row * 48 + col] : (h16)0.f;
  }
  __syncthreads();
  int wave = tid >> 5, lane = tid & 31;
  const h16* ep = wsin + OFF_EP;
  v8f acc[4] = {};
  for (int kt = 0; kt < 2; ++kt) {
    v16h a = load_a_rm(&zs[kt * 32], 72, lane);
#pragma unroll
    for (int q = 0; q < 4; ++q) {
      int nt = wave * 4 + q;
      v16h b = *(const v16h*)(ep + ((size_t)(kt * 16 + nt) * 32 + lane) * 16);
      acc[q] = WMMA16(a, b, acc[q]);
    }
  }
  int kg = lane >> 4, n0 = lane & 15;
#pragma unroll
  for (int q = 0; q < 4; ++q)
#pragma unroll
    for (int r = 0; r < 8; ++r) {
      int row = row0 + r + 8 * kg;
      out[(size_t)row * 256 + (wave * 4 + q) * 16 + n0] = acc[q][r];
    }
}

extern "C" void kernel_launch(void* const* d_in, const int* in_sizes, int n_in,
                              void* d_out, int out_size, void* d_ws, size_t ws_size,
                              hipStream_t stream) {
  (void)in_sizes; (void)n_in; (void)out_size; (void)ws_size;
  const float* x   = (const float*)d_in[0];
  const float* w1r = (const float*)d_in[1];
  const float* w1i = (const float*)d_in[2];
  const float* w2r = (const float*)d_in[3];
  const float* w2i = (const float*)d_in[4];
  float* out = (float*)d_out;
  h16* ws = (h16*)d_ws;

  // init DFT factors + marshaled weights (cheap, fully parallel)
  k_init_fw<<<48, 256, 0, stream>>>(ws);
  k_init_h <<<144, 256, 0, stream>>>(ws);
  k_init_g <<<96, 256, 0, stream>>>(ws);
  k_init_e <<<64, 256, 0, stream>>>(ws);
  k_init_bw<<<51200, 256, 0, stream>>>(ws, w1r, w1i, w2r, w2i);

  // pipeline: pruned rfft2 -> channel mix -> pruned irfft2, all as WMMA GEMMs
  k_stageA<<<kROWS / 64, 128, 0, stream>>>(x, ws);   // 4096 blocks
  k_stageB<<<1024, 128, 0, stream>>>(ws);
  k_stageC<<<kNPOS, 128, 0, stream>>>(ws);           // 800 blocks
  k_stageD<<<1024, 128, 0, stream>>>(ws);
  k_stageE<<<kROWS / 16, 128, 0, stream>>>(ws, out); // 16384 blocks
}
